// DynamicRNNLayer_20495583937036
// MI455X (gfx1250) — matrix operand order repaired and checked by reading, not compiled, not run on hardware
//
#include <hip/hip_runtime.h>
#include <hip/hip_bf16.h>

typedef __attribute__((ext_vector_type(16))) __bf16 v16bf;
typedef __attribute__((ext_vector_type(8)))  __bf16 v8bf;
typedef __attribute__((ext_vector_type(8)))  float  v8f;
typedef __attribute__((ext_vector_type(4)))  unsigned int v4u;
typedef __attribute__((ext_vector_type(8)))  int v8i32;
typedef __attribute__((ext_vector_type(4)))  int v4i32;

constexpr int Bc   = 64;
constexpr int Tc   = 256;
constexpr int Dc   = 512;
constexpr int Hc   = 1024;
constexpr int KTOT = Dc + Hc;       // 1536 = concat(x, h) feature dim
constexpr int G4H  = 4 * Hc;        // 4096 gate columns (i, j, f, o)
constexpr int NKB  = KTOT / 32;     // 48 K-blocks of 32 (bf16 WMMA K)
constexpr int GATE_STRIDE = 64 * NKB * 512;  // elements between gate regions in wpack

// ---- workspace layout (bytes) ----
constexpr size_t WS_WPACK = 0;                                   // 12,582,912
constexpr size_t WS_XBF   = WS_WPACK + (size_t)KTOT * G4H * 2;   // 16,777,216
constexpr size_t WS_H0    = WS_XBF + (size_t)Bc * Tc * Dc * 2;   // 131,072
constexpr size_t WS_H1    = WS_H0 + (size_t)Bc * Hc * 2;
constexpr size_t WS_C     = WS_H1 + (size_t)Bc * Hc * 2;         // 262,144

__device__ __forceinline__ __bf16 f2bf(float f) {
  union { float f; unsigned u; } v; v.f = f;
  unsigned r = (v.u + 0x7FFFu + ((v.u >> 16) & 1u)) >> 16;
  unsigned short s = (unsigned short)r;
  __bf16 out; __builtin_memcpy(&out, &s, sizeof(out));
  return out;
}

__device__ __forceinline__ float sigm(float x) {
  return 1.0f / (1.0f + __expf(-x));
}
__device__ __forceinline__ float tanh_fast(float x) {
  float e = __expf(-2.0f * x);
  return (1.0f - e) / (1.0f + e);
}

// TDM: load a 512x4 bf16 tile (512 elements per gate, 4 gates, row stride =
// GATE_STRIDE elements) from packed W into LDS at lds_byte (4KB contiguous).
__device__ __forceinline__ void tdm_load_b_tile(const __bf16* gptr, unsigned lds_byte) {
  unsigned long long ga = (unsigned long long)(size_t)gptr;
  v4u g0;
  g0[0] = 1u;                                              // count=1, user mode
  g0[1] = lds_byte;                                        // D#.lds_addr
  g0[2] = (unsigned)(ga & 0xFFFFFFFFu);                    // global_addr[31:0]
  g0[3] = (unsigned)((ga >> 32) & 0x1FFFFFFu) | (2u << 30);// addr[56:32] | type=2
  v8i32 g1;
  g1[0] = (1 << 16);                 // workgroup_mask=0, data_size=1 (2 bytes)
  g1[1] = (int)(512u << 16);         // tensor_dim0[15:0] = 512 at bits 63:48
  g1[2] = (int)(4u << 16);           // tensor_dim0[31:16]=0 | tensor_dim1[15:0]=4
  g1[3] = (int)(512u << 16);         // tensor_dim1[31:16]=0 | tile_dim0=512
  g1[4] = 4;                         // tile_dim1=4 (gates), tile_dim2=0
  g1[5] = GATE_STRIDE;               // tensor_dim0_stride[31:0]
  g1[6] = 0;                         // stride0 hi, stride1 lo
  g1[7] = 0;                         // stride1 hi
  v4i32 z4 = {0, 0, 0, 0};
  v8i32 z8 = {0, 0, 0, 0, 0, 0, 0, 0};
  __builtin_amdgcn_tensor_load_to_lds(g0, g1, z4, z4, z8, 0);
}

// Pack W [KTOT, 4H] f32 row-major -> bf16 WMMA B-fragment order.
// Lane L of a packed 32x16 (KxN) B tile holds column L&15, K = (L>>4)*16 + e (code's own order, same as A).
__global__ __launch_bounds__(256)
void pack_w_kernel(const float* __restrict__ W, __bf16* __restrict__ wpack) {
  int p = blockIdx.x * 256 + threadIdx.x;      // 0 .. 6,291,455
  int blk  = p >> 9;
  int rem  = p & 511;
  int lane = rem >> 4;
  int e    = rem & 15;
  int ublk = blk / NKB;
  int kb   = blk - ublk * NKB;
  int k    = kb * 32 + (lane >> 4) * 16 + e;
  int col  = (ublk << 4) + (lane & 15);
  wpack[p] = f2bf(W[(size_t)k * G4H + col]);
}

__global__ __launch_bounds__(256)
void convert_x_kernel(const float* __restrict__ x, __bf16* __restrict__ xbf) {
  size_t i = (size_t)blockIdx.x * 256 + threadIdx.x;  // 8,388,608 elements
  xbf[i] = f2bf(x[i]);
}

__global__ __launch_bounds__(256)
void init_state_kernel(float* __restrict__ c, __bf16* __restrict__ h0,
                       float* __restrict__ h_out) {
  int i = blockIdx.x * 256 + threadIdx.x;             // 65,536 elements
  c[i] = 0.0f;
  h0[i] = f2bf(0.0f);
  h_out[i] = 0.0f;
}

// One fused LSTM timestep. Grid: 64 blocks x 128 threads.
// utile = blockIdx.x (16 hidden units), mtile = wave (16 batch rows).
// Wave 0 stages the shared B operand (4 gates x 32x16 bf16 = 4KB per K-block)
// into an LDS double buffer with the Tensor Data Mover; all 4 waves consume it
// via ds_load_b128 and run 4 v_wmma_f32_16x16x32_bf16 per K-block.
__global__ __launch_bounds__(128)
void lstm_step_kernel(const float* __restrict__ bias,
                      const int* __restrict__ seq_len,
                      const __bf16* __restrict__ xbf,
                      const __bf16* __restrict__ wpack,
                      const __bf16* __restrict__ h_prev,
                      __bf16* __restrict__ h_next,
                      float* __restrict__ c_state,
                      float* __restrict__ h_out,
                      int t) {
  __shared__ __attribute__((aligned(32))) __bf16 smemB[2][4 * 512];  // 8KB

  const int lane  = threadIdx.x & 31;
  const int wave  = threadIdx.x >> 5;     // 0..3 = mtile
  const int utile = blockIdx.x;           // 0..63
  const int mbase = wave << 4;
  const int ubase = utile << 4;
  const int khalf = lane >> 4;
  const int lcol  = lane & 15;
  const int u     = ubase + lcol;

  // Accumulators start at the bias (broadcast down the 16 rows).
  v8f acc0, acc1, acc2, acc3;
  {
    float b0 = bias[0 * Hc + u], b1 = bias[1 * Hc + u];
    float b2 = bias[2 * Hc + u], b3 = bias[3 * Hc + u];
    acc0 = (v8f){b0, b0, b0, b0, b0, b0, b0, b0};
    acc1 = (v8f){b1, b1, b1, b1, b1, b1, b1, b1};
    acc2 = (v8f){b2, b2, b2, b2, b2, b2, b2, b2};
    acc3 = (v8f){b3, b3, b3, b3, b3, b3, b3, b3};
  }

  // A-fragment: lane L supplies row (mbase + L&15); K-half chosen by L>>4.
  const int arow = mbase + lcol;
  const __bf16* xrow = xbf + ((size_t)arow * Tc + t) * Dc;
  const __bf16* hrow = h_prev + (size_t)arow * Hc;

  // Gate-0 source for K-block kb; gates 1..3 at +GATE_STRIDE each (TDM rows).
  const __bf16* wsrc = wpack + ((size_t)utile * NKB << 9);
  const unsigned lds0 = (unsigned)(size_t)&smemB[0][0];
  const unsigned lds1 = (unsigned)(size_t)&smemB[1][0];

  // Prologue: stage K-block 0 into buffer 0.
  if (wave == 0) tdm_load_b_tile(wsrc, lds0);

  for (int kb = 0; kb < NKB; ++kb) {
    if (wave == 0) __builtin_amdgcn_s_wait_tensorcnt(0);
    __syncthreads();  // buffer kb visible to all waves

    // Stage next K-block into the other buffer while we compute this one.
    if (wave == 0 && (kb + 1) < NKB)
      tdm_load_b_tile(wsrc + ((size_t)(kb + 1) << 9), (kb & 1) ? lds0 : lds1);

    // K-chunks 0..15 come from x_t, 16..47 from h_prev (Dc = 16*32).
    const __bf16* ap = (kb < 16) ? (xrow + kb * 32) : (hrow + (kb - 16) * 32);
    v8bf alo = *(const v8bf*)(ap + khalf * 16);
    v8bf ahi = *(const v8bf*)(ap + khalf * 16 + 8);
    v16bf afrag = __builtin_shufflevector(
        alo, ahi, 0, 1, 2, 3, 4, 5, 6, 7, 8, 9, 10, 11, 12, 13, 14, 15);

    const __bf16* lb = &smemB[kb & 1][0];
    #pragma unroll
    for (int g = 0; g < 4; ++g) {
      const __bf16* bp = lb + (g << 9) + (lane << 4);
      v8bf blo = *(const v8bf*)(bp);
      v8bf bhi = *(const v8bf*)(bp + 8);
      v16bf bfrag = __builtin_shufflevector(
          blo, bhi, 0, 1, 2, 3, 4, 5, 6, 7, 8, 9, 10, 11, 12, 13, 14, 15);
      if (g == 0)
        acc0 = __builtin_amdgcn_wmma_f32_16x16x32_bf16(
            false, afrag, false, bfrag, (short)0, acc0, false, false);
      else if (g == 1)
        acc1 = __builtin_amdgcn_wmma_f32_16x16x32_bf16(
            false, afrag, false, bfrag, (short)0, acc1, false, false);
      else if (g == 2)
        acc2 = __builtin_amdgcn_wmma_f32_16x16x32_bf16(
            false, afrag, false, bfrag, (short)0, acc2, false, false);
      else
        acc3 = __builtin_amdgcn_wmma_f32_16x16x32_bf16(
            false, afrag, false, bfrag, (short)0, acc3, false, false);
    }
  }

  // Epilogue. C/D layout: lane L holds column L&15, rows (L>>4)*8 + v.
  const int rbase = mbase + khalf * 8;
  #pragma unroll
  for (int v = 0; v < 8; ++v) {
    const int brow = rbase + v;
    const int idx  = brow * Hc + u;
    const int sl   = seq_len[brow];
    float cold = c_state[idx];
    float zi = acc0[v], zj = acc1[v], zf = acc2[v], zo = acc3[v];
    float nc = cold * sigm(zf + 1.0f) + sigm(zi) * tanh_fast(zj);  // forget bias 1.0
    float nh = tanh_fast(nc) * sigm(zo);
    bool m = (t < sl);                       // dynamic_rnn copy-through
    if (m) {
      c_state[idx] = nc;
      h_out[idx]   = nh;                     // final h accumulates in d_out
    }
    h_next[idx] = m ? f2bf(nh) : h_prev[idx];
  }
}

extern "C" void kernel_launch(void* const* d_in, const int* in_sizes, int n_in,
                              void* d_out, int out_size, void* d_ws, size_t ws_size,
                              hipStream_t stream) {
  const float* x       = (const float*)d_in[0];
  const int*   seq_len = (const int*)d_in[1];
  const float* W       = (const float*)d_in[2];
  const float* bias    = (const float*)d_in[3];
  float* h_out = (float*)d_out;

  char* ws = (char*)d_ws;
  __bf16* wpack = (__bf16*)(ws + WS_WPACK);
  __bf16* xbf   = (__bf16*)(ws + WS_XBF);
  __bf16* h0    = (__bf16*)(ws + WS_H0);
  __bf16* h1    = (__bf16*)(ws + WS_H1);
  float*  cst   = (float*)(ws + WS_C);

  // One-time (per call) prep: W -> packed bf16 B-fragments, x -> bf16, zero state.
  pack_w_kernel<<<(KTOT * G4H) / 256, 256, 0, stream>>>(W, wpack);
  convert_x_kernel<<<(Bc * Tc * Dc) / 256, 256, 0, stream>>>(x, xbf);
  init_state_kernel<<<(Bc * Hc) / 256, 256, 0, stream>>>(cst, h0, h_out);

  // Recurrence: stream-ordered launches provide the step-to-step barrier;
  // h is double-buffered in bf16 so readers never race writers within a step.
  for (int t = 0; t < Tc; ++t) {
    const __bf16* hp = (t & 1) ? h1 : h0;
    __bf16*       hn = (t & 1) ? h0 : h1;
    lstm_step_kernel<<<64, 128, 0, stream>>>(bias, seq_len, xbf, wpack,
                                             hp, hn, cst, h_out, t);
  }
}
